// FastTabNet_59176059404352
// MI455X (gfx1250) — matrix operand order repaired and checked by reading, not compiled, not run on hardware
//
#include <hip/hip_runtime.h>
#include <hip/hip_bf16.h>
#include <math.h>

// ---------------------------------------------------------------------------
// Fully-fused TabNet forward for MI455X (gfx1250, wave32, WMMA).
//
// Round-4 change: GEMM loops process 2 column-tile pairs per outer iteration
// (4 f32 accumulators live) so every A-fragment ds_load_b128 feeds 4 WMMAs
// instead of 2 -> hot-loop LDS traffic halved.
// ---------------------------------------------------------------------------

typedef __attribute__((ext_vector_type(16))) _Float16 v16h;
typedef __attribute__((ext_vector_type(8)))  _Float16 v8h;
typedef __attribute__((ext_vector_type(8)))  float    v8f;

#define TB      128      // threads per block (4 wave32)
#define ROWS    16       // batch rows per block
#define BATCH   65536
#define NSTEPS  5
#define BN_EPS_C 1e-5f
#define SCALE_C  0.70710678118654752440f
#define GAMMA_C  1.5f

// ---- WMMA fragment loaders -------------------------------------------------
// A (16x32 f16, MxK): lanes 0-15 -> M=lane, kb=0; lanes 16-31 -> M=lane-16,
// kb=8. Elements 0..7 = K kb..kb+7 ; elements 8..15 = K kb+16..kb+23.
__device__ __forceinline__ v16h load_a_frag(const _Float16* A, int k0, int ldm) {
    int lane = threadIdx.x & 31;
    int m    = lane & 15;
    int kb   = k0 + ((lane >> 4) << 3);
    const _Float16* p = A + m * ldm + kb;
    v8h lo = *(const v8h*)(p);        // ds_load_b128
    v8h hi = *(const v8h*)(p + 16);   // ds_load_b128
    v16h r;
#pragma unroll
    for (int e = 0; e < 8; ++e) { r[e] = lo[e]; r[8 + e] = hi[e]; }
    return r;
}

// B (32x16 f16, KxN) pre-packed: lane's 16 f16 are 32 contiguous bytes.
__device__ __forceinline__ v16h load_b_frag(const _Float16* __restrict__ Wp,
                                            int kc, int jt, int ntiles) {
    int lane = threadIdx.x & 31;
    const _Float16* p = Wp + ((((size_t)kc * ntiles + jt) * 32 + lane) << 4);
    return *(const v16h*)p;           // global_load_b128 x2
}

__device__ __forceinline__ v8f wmma16(v16h a, v16h b, v8f c) {
    return __builtin_amdgcn_wmma_f32_16x16x32_f16(false, a, false, b,
                                                  (short)0, c, false, false);
}

__device__ __forceinline__ float fast_sigmoid(float x) {
    return __builtin_amdgcn_rcpf(1.0f + __expf(-x));   // v_rcp_f32 + v_exp_f32
}

// ---- GLU epilogue for one 16x16 tile pair ---------------------------------
__device__ __forceinline__ void glu_epilogue(const _Float16* Ain,
                                             _Float16* Aout,
                                             const float* __restrict__ bnp,
                                             int j, int lane, int nloc,
                                             v8f acc_a, v8f acc_b,
                                             bool residual) {
    int ca = j * 16 + nloc;           // GLU "a" column
    int cb = 256 + ca;                // GLU "b" (gate) column
    float sa = bnp[ca], ta = bnp[512 + ca];
    float sb = bnp[cb], tb = bnp[512 + cb];
#pragma unroll
    for (int r = 0; r < 8; ++r) {
        int m = r + ((lane >> 4) << 3);          // C/D layout row
        float ha = acc_a[r] * sa + ta;
        float hb = acc_b[r] * sb + tb;
        float g  = ha * fast_sigmoid(hb);
        float ov = residual
                 ? ((float)Ain[m * 256 + ca] + g) * SCALE_C
                 : g;
        Aout[m * 256 + ca] = (_Float16)ov;
    }
}

// ---- one GLU layer: [16,256] @ [256,512] -> BN -> GLU -> (residual) --------
// bnp: precomputed [2][512] = {scale[512], shift[512]}
__device__ void glu_layer(const _Float16* Ain, _Float16* Aout,
                          const _Float16* __restrict__ Wp,
                          const float* __restrict__ bnp,
                          bool residual) {
    int wave = threadIdx.x >> 5;
    int lane = threadIdx.x & 31;
    int nloc = lane & 15;
#pragma unroll 1
    for (int jj = 0; jj < 2; ++jj) {
        int j0 = wave * 4 + jj * 2;       // this iteration: pairs j0, j0+1
        v8f acc_a0 = {}, acc_b0 = {};
        v8f acc_a1 = {}, acc_b1 = {};
#pragma unroll 4
        for (int k = 0; k < 8; ++k) {     // K = 256 in chunks of 32
            v16h af  = load_a_frag(Ain, k * 32, 256);
            v16h b0a = load_b_frag(Wp, k, j0,          32);
            v16h b0b = load_b_frag(Wp, k, 16 + j0,     32);
            v16h b1a = load_b_frag(Wp, k, j0 + 1,      32);
            v16h b1b = load_b_frag(Wp, k, 16 + j0 + 1, 32);
            acc_a0 = wmma16(af, b0a, acc_a0);
            acc_b0 = wmma16(af, b0b, acc_b0);
            acc_a1 = wmma16(af, b1a, acc_a1);
            acc_b1 = wmma16(af, b1b, acc_b1);
        }
        glu_epilogue(Ain, Aout, bnp, j0,     lane, nloc, acc_a0, acc_b0, residual);
        glu_epilogue(Ain, Aout, bnp, j0 + 1, lane, nloc, acc_a1, acc_b1, residual);
    }
    __syncthreads();
}

// ---------------------------------------------------------------------------
__global__ __launch_bounds__(TB) void tabnet_fused(
    const float* __restrict__ x,            // [B,256]
    const float* __restrict__ init_bnp,     // [2][256] scale/shift
    const _Float16* __restrict__ sW0p,      // packed 256x512
    const _Float16* __restrict__ sW1p,      // packed 256x512
    const float* __restrict__ init_sbnp,    // [2][2][512]
    const _Float16* __restrict__ init_iWp,  // packed 2x(256x512)
    const float* __restrict__ init_ibnp,    // [2][2][512]
    const float* __restrict__ step_sbnp,    // [10][2][512]
    const _Float16* __restrict__ step_iWp,  // packed 10x(256x512)
    const float* __restrict__ step_ibnp,    // [10][2][512]
    const _Float16* __restrict__ attn_Wp,   // packed 5x(128x256)
    const float* __restrict__ attn_bnp,     // [5][2][256]
    const float* __restrict__ fc_W,         // [128,10]
    const float* __restrict__ fc_b,         // [10]
    float* __restrict__ logits,             // [B,10]
    float* __restrict__ loss,               // [1]
    float* __restrict__ masks)              // [B,5,256]
{
    // LDS: ping-pong f16 activations (16 KB) unioned with f32 attention
    // scores Z (16 KB); plus x_bn(f16), prior(f32), h_a(f16), agg(f32).
    __shared__ __attribute__((aligned(16))) _Float16 sAbuf[2][ROWS * 256];
    __shared__ __attribute__((aligned(16))) _Float16 sXbn[ROWS * 256];
    __shared__ float    sPrior[ROWS * 256];
    __shared__ __attribute__((aligned(16))) _Float16 sHa[ROWS * 128];
    __shared__ float    sAgg[ROWS * 128];

    _Float16* A0 = sAbuf[0];
    _Float16* A1 = sAbuf[1];
    float*    Z  = (float*)&sAbuf[0][0];        // 16*256 f32 == both A buffers

    const int grow = blockIdx.x * ROWS;
    float entAcc = 0.0f;

    // ---- init: x_bn, prior=1 ----------------------------------------------
#pragma unroll 1
    for (int i = 0; i < 32; ++i) {
        int idx = threadIdx.x + TB * i;         // 0..4095 == m*256+c
        int m = idx >> 8, c = idx & 255;
        float v = x[(size_t)(grow + m) * 256 + c] * init_bnp[c]
                + init_bnp[256 + c];
        sXbn[idx]   = (_Float16)v;
        A0[idx]     = (_Float16)v;
        sPrior[idx] = 1.0f;
    }
    __syncthreads();

    // ---- init feature transform (4 GLU layers, result lands in A0) --------
    glu_layer(A0, A1, sW0p, init_sbnp,                     false);
    glu_layer(A1, A0, sW1p, init_sbnp + 1024,              true);
    glu_layer(A0, A1, init_iWp,             init_ibnp,         true);
    glu_layer(A1, A0, init_iWp + 256 * 512, init_ibnp + 1024,  true);

    // h_a = h[:,128:], agg = 0
#pragma unroll 1
    for (int i = 0; i < 16; ++i) {
        int idx = threadIdx.x + TB * i;         // 0..2047 == m*128+c
        int m = idx >> 7, c = idx & 127;
        sAgg[idx] = 0.0f;
        sHa[idx]  = A0[m * 256 + 128 + c];
    }
    __syncthreads();

    // ---- decision steps ----------------------------------------------------
#pragma unroll 1
    for (int s = 0; s < NSTEPS; ++s) {
        // attention: Z = bn(h_a @ attn_W[s]) * prior   (K=128, N=256)
        {
            const _Float16* awp = attn_Wp  + (size_t)s * 128 * 256;
            const float*    abn = attn_bnp + (size_t)s * 512;
            int wave = threadIdx.x >> 5;
            int lane = threadIdx.x & 31;
            int nloc = lane & 15;
#pragma unroll 1
            for (int jj = 0; jj < 2; ++jj) {
                int j0 = wave * 4 + jj * 2;
                v8f acc0 = {}, acc1 = {};
#pragma unroll
                for (int k = 0; k < 4; ++k) {
                    v16h af = load_a_frag(sHa, k * 32, 128);
                    v16h b0 = load_b_frag(awp, k, j0,     16);
                    v16h b1 = load_b_frag(awp, k, j0 + 1, 16);
                    acc0 = wmma16(af, b0, acc0);
                    acc1 = wmma16(af, b1, acc1);
                }
#pragma unroll
                for (int t = 0; t < 2; ++t) {
                    int c = (j0 + t) * 16 + nloc;
                    v8f acc = t ? acc1 : acc0;
                    float sc = abn[c], tc = abn[256 + c];
#pragma unroll
                    for (int r = 0; r < 8; ++r) {
                        int m = r + ((lane >> 4) << 3);
                        Z[m * 256 + c] =
                            (acc[r] * sc + tc) * sPrior[m * 256 + c];
                    }
                }
            }
            __syncthreads();
        }

        // sparsemax per row (8 lanes/row, 32 elems/lane, tau bisection)
        {
            int row = threadIdx.x >> 3;
            int sub = threadIdx.x & 7;
            float z[32];
#pragma unroll
            for (int i = 0; i < 32; ++i) z[i] = Z[row * 256 + sub + 8 * i];
            __syncthreads();                    // Z fully consumed; A0 reusable

            float mx = z[0];
#pragma unroll
            for (int i = 1; i < 32; ++i) mx = fmaxf(mx, z[i]);
            for (int off = 4; off > 0; off >>= 1)
                mx = fmaxf(mx, __shfl_xor(mx, off, 8));

            float lo = mx - 1.0f, hi = mx;      // tau* in [mx-1, mx)
#pragma unroll 1
            for (int it = 0; it < 30; ++it) {
                float tau = 0.5f * (lo + hi);
                float sm = 0.0f;
#pragma unroll
                for (int i = 0; i < 32; ++i) sm += fmaxf(z[i] - tau, 0.0f);
                for (int off = 4; off > 0; off >>= 1)
                    sm += __shfl_xor(sm, off, 8);
                if (sm > 1.0f) lo = tau; else hi = tau;
            }
            float tau = 0.5f * (lo + hi);

            float ent = 0.0f;
#pragma unroll 1
            for (int i = 0; i < 32; ++i) {
                int   c = sub + 8 * i;
                float a = fmaxf(z[i] - tau, 0.0f);
                masks[((size_t)(grow + row) * NSTEPS + s) * 256 + c] = a;
                ent += -a * __logf(a + 1e-15f);
                sPrior[row * 256 + c] *= (GAMMA_C - a);
                A0[row * 256 + c] =
                    (_Float16)(a * (float)sXbn[row * 256 + c]);
            }
            entAcc += ent;
            __syncthreads();
        }

        // masked feature transform (result -> A0)
        glu_layer(A0, A1, sW0p, step_sbnp + (size_t)(s * 2    ) * 1024, false);
        glu_layer(A1, A0, sW1p, step_sbnp + (size_t)(s * 2 + 1) * 1024, true);
        glu_layer(A0, A1, step_iWp + (size_t)(s * 2    ) * 256 * 512,
                           step_ibnp + (size_t)(s * 2    ) * 1024, true);
        glu_layer(A1, A0, step_iWp + (size_t)(s * 2 + 1) * 256 * 512,
                           step_ibnp + (size_t)(s * 2 + 1) * 1024, true);

        // agg += relu(h[:,:128]);  h_a = h[:,128:]
#pragma unroll 1
        for (int i = 0; i < 16; ++i) {
            int idx = threadIdx.x + TB * i;
            int m = idx >> 7, c = idx & 127;
            float v = (float)A0[m * 256 + c];
            sAgg[idx] += fmaxf(v, 0.0f);
            sHa[idx]   = A0[m * 256 + 128 + c];
        }
        __syncthreads();
    }

    // ---- logits: [16,128] @ [128,10] + b -----------------------------------
    if (threadIdx.x < ROWS * 10) {
        int m = threadIdx.x / 10;
        int c = threadIdx.x - m * 10;
        float acc = fc_b[c];
#pragma unroll 1
        for (int k = 0; k < 128; ++k)
            acc += sAgg[m * 128 + k] * fc_W[k * 10 + c];
        logits[(size_t)(grow + m) * 10 + c] = acc;
    }

    // ---- sparsity loss: block reduce + one atomicAdd -----------------------
    float* red = Z;                             // Z region free now
    red[threadIdx.x] = entAcc;
    __syncthreads();
    if (threadIdx.x == 0) {
        float sm = 0.0f;
        for (int i = 0; i < TB; ++i) sm += red[i];
        atomicAdd(loss, sm * (1.0f / ((float)BATCH * (float)NSTEPS)));
    }
}

// ---- weight pre-pack: f32 row-major [K,N] -> f16 B-fragment order ----------
__global__ void tabnet_pack_w(const float* __restrict__ W,
                              _Float16* __restrict__ out, int K, int N) {
    size_t i = (size_t)blockIdx.x * 256 + threadIdx.x;
    if (i >= (size_t)K * N) return;
    int e    = (int)(i & 15);
    int lane = (int)((i >> 4) & 31);
    size_t rest = i >> 9;                 // kc*(N/16) + jt
    int nt = N >> 4;
    int jt = (int)(rest % nt);
    int kc = (int)(rest / nt);
    int n  = jt * 16 + (lane & 15);
    int k  = kc * 32 + ((lane >> 4) << 3) + (e & 7) + ((e & 8) << 1);
    out[i] = (_Float16)W[(size_t)k * N + n];
}

// ---- BN fold: [4][C] (gamma,beta,mean,var) -> [2][C] (scale,shift) ---------
__global__ void tabnet_pack_bn(const float* __restrict__ bn,
                               float* __restrict__ out, int C) {
    int c = blockIdx.x * 256 + threadIdx.x;
    if (c >= C) return;
    float s = bn[c] * rsqrtf(bn[3 * C + c] + BN_EPS_C);
    out[c]     = s;
    out[C + c] = bn[C + c] - bn[2 * C + c] * s;
}

__global__ void tabnet_zero_loss(float* p) { *p = 0.0f; }

// ---------------------------------------------------------------------------
extern "C" void kernel_launch(void* const* d_in, const int* in_sizes, int n_in,
                              void* d_out, int out_size, void* d_ws, size_t ws_size,
                              hipStream_t stream) {
    (void)in_sizes; (void)n_in; (void)out_size; (void)ws_size;

    const float* x         = (const float*)d_in[0];
    const float* init_bn   = (const float*)d_in[1];
    const float* sW0       = (const float*)d_in[2];
    const float* sW1       = (const float*)d_in[3];
    const float* init_sbn  = (const float*)d_in[4];
    const float* init_iW   = (const float*)d_in[5];
    const float* init_ibn  = (const float*)d_in[6];
    const float* step_sbn  = (const float*)d_in[7];
    const float* step_iW   = (const float*)d_in[8];
    const float* step_ibn  = (const float*)d_in[9];
    const float* attn_W    = (const float*)d_in[10];
    const float* attn_bn   = (const float*)d_in[11];
    const float* fc_W      = (const float*)d_in[12];
    const float* fc_b      = (const float*)d_in[13];

    float* out    = (float*)d_out;
    float* logits = out;                                  // B*10
    float* loss   = out + (size_t)BATCH * 10;             // 1
    float* masks  = loss + 1;                             // B*5*256

    // ---- workspace layout --------------------------------------------------
    const size_t BIGW  = 256 * 512;                       // 131072 halves
    const size_t ATTNW = 128 * 256;                       // 32768 halves
    _Float16* wsp      = (_Float16*)d_ws;
    _Float16* sW0p     = wsp;                 // 1
    _Float16* sW1p     = sW0p + BIGW;         // 1
    _Float16* init_iWp = sW1p + BIGW;         // 2
    _Float16* step_iWp = init_iWp + 2 * BIGW; // 10
    _Float16* attn_Wp  = step_iWp + 10 * BIGW;// 5
    // fp32 BN (scale,shift) after the f16 weights (8B-aligned)
    float* bnbase     = (float*)(attn_Wp + 5 * ATTNW);
    float* init_sbnp  = bnbase;               //  2 * 1024
    float* init_ibnp  = init_sbnp + 2 * 1024; //  2 * 1024
    float* step_sbnp  = init_ibnp + 2 * 1024; // 10 * 1024
    float* step_ibnp  = step_sbnp + 10 * 1024;// 10 * 1024
    float* attn_bnp   = step_ibnp + 10 * 1024;//  5 * 512
    float* init_bnp   = attn_bnp + 5 * 512;   //  1 * 512   (~4.1 MB total)

    const int PT = 256;
    const int gBig  = (int)((BIGW  + PT - 1) / PT);
    const int gAttn = (int)((ATTNW + PT - 1) / PT);

    tabnet_pack_w<<<gBig, PT, 0, stream>>>(sW0, sW0p, 256, 512);
    tabnet_pack_w<<<gBig, PT, 0, stream>>>(sW1, sW1p, 256, 512);
    for (int j = 0; j < 2; ++j)
        tabnet_pack_w<<<gBig, PT, 0, stream>>>(init_iW + (size_t)j * BIGW,
                                               init_iWp + (size_t)j * BIGW,
                                               256, 512);
    for (int j = 0; j < 10; ++j)
        tabnet_pack_w<<<gBig, PT, 0, stream>>>(step_iW + (size_t)j * BIGW,
                                               step_iWp + (size_t)j * BIGW,
                                               256, 512);
    for (int j = 0; j < 5; ++j)
        tabnet_pack_w<<<gAttn, PT, 0, stream>>>(attn_W + (size_t)j * ATTNW,
                                                attn_Wp + (size_t)j * ATTNW,
                                                128, 256);

    for (int j = 0; j < 2; ++j)
        tabnet_pack_bn<<<2, PT, 0, stream>>>(init_sbn + (size_t)j * 2048,
                                             init_sbnp + (size_t)j * 1024, 512);
    for (int j = 0; j < 2; ++j)
        tabnet_pack_bn<<<2, PT, 0, stream>>>(init_ibn + (size_t)j * 2048,
                                             init_ibnp + (size_t)j * 1024, 512);
    for (int j = 0; j < 10; ++j)
        tabnet_pack_bn<<<2, PT, 0, stream>>>(step_sbn + (size_t)j * 2048,
                                             step_sbnp + (size_t)j * 1024, 512);
    for (int j = 0; j < 10; ++j)
        tabnet_pack_bn<<<2, PT, 0, stream>>>(step_ibn + (size_t)j * 2048,
                                             step_ibnp + (size_t)j * 1024, 512);
    for (int j = 0; j < 5; ++j)
        tabnet_pack_bn<<<1, PT, 0, stream>>>(attn_bn + (size_t)j * 1024,
                                             attn_bnp + (size_t)j * 512, 256);
    tabnet_pack_bn<<<1, PT, 0, stream>>>(init_bn, init_bnp, 256);

    tabnet_zero_loss<<<1, 1, 0, stream>>>(loss);
    tabnet_fused<<<BATCH / ROWS, TB, 0, stream>>>(
        x, init_bnp, sW0p, sW1p, init_sbnp, init_iWp, init_ibnp,
        step_sbnp, step_iWp, step_ibnp, attn_Wp, attn_bnp, fc_W, fc_b,
        logits, loss, masks);
}
